// DGCNNSeg_29033978921716
// MI455X (gfx1250) — matrix operand order repaired and checked by reading, not compiled, not run on hardware
//
#include <hip/hip_runtime.h>

// ---------------------------------------------------------------------------
// DGCNN part-seg forward for MI455X (gfx1250, wave32).
//   - KNN pairwise-distance GEMM: V_WMMA_F32_16X16X4_F32 (exact f32)
//   - All channel-mixing GEMMs:   V_WMMA_F32_16X16X32_F16 (f16 in, f32 acc)
//   - BN (training stats) + LReLU as streaming passes, max-over-k/n fused.
// Round 3: BN stats -> deterministic two-stage reduction (grid scales with
// tensor size, not channel count); bn_lrelu vectorized to 8xf16.
// ---------------------------------------------------------------------------

typedef __attribute__((ext_vector_type(16))) _Float16 v16h;
typedef __attribute__((ext_vector_type(8)))  _Float16 v8h;
typedef __attribute__((ext_vector_type(8)))  float    v8f;
typedef __attribute__((ext_vector_type(2)))  float    v2f;

#define NPTS   4096
#define NB     8
#define KNN_K  20
#define ECOLS  (NPTS * KNN_K)   // 81920 edge columns per batch
#define STCH   2048             // BN stats column chunk

enum { MODE_F16 = 0, MODE_F32 = 1, MODE_CAT = 2, MODE_EDGE = 3 };

__device__ __forceinline__ float lrelu(float v) { return v >= 0.f ? v : 0.2f * v; }

// ---------------------------------------------------------------------------
// squared norms per point:  xsq[b][n] = sum_c x[b][c][n]^2
// ---------------------------------------------------------------------------
template<int C>
__global__ __launch_bounds__(256)
void sqnorm_kernel(const float* __restrict__ x, long sB, int N,
                   float* __restrict__ xsq)
{
    int n = blockIdx.x * 256 + threadIdx.x;
    int b = blockIdx.y;
    if (n >= N) return;
    const float* xb = x + (long)b * sB;
    float s = 0.f;
#pragma unroll 4
    for (int c = 0; c < C; ++c) { float v = xb[(long)c * N + n]; s += v * v; }
    xsq[(long)b * N + n] = s;
}

// ---------------------------------------------------------------------------
// KNN: top-20 of pd[i][j] = 2*<x_i,x_j> - |x_i|^2 - |x_j|^2 per row.
// Block = 128 threads (4 waves), covers 16 rows; each wave walks disjoint
// 16-column tiles, inner products via f32 WMMA, tile staged through LDS,
// 2 lanes per row keep sorted top-20 lists; final 8-way merge per row.
// ---------------------------------------------------------------------------
template<int CP, int C>   // CP: padded channels (mult of 4); C: real channels
__global__ __launch_bounds__(128)
void knn_kernel(const float* __restrict__ x, long sB, int N,
                const float* __restrict__ xsq, int* __restrict__ idxOut)
{
    int b    = blockIdx.y;
    int r0   = blockIdx.x * 16;
    int tid  = threadIdx.x;
    int wv   = tid >> 5;
    int lane = tid & 31;
    int l15  = lane & 15;
    int hi   = lane >> 4;

    __shared__ float tbuf[4][16][17];   // per-wave pd tile staging
    __shared__ float mV[16][160];
    __shared__ int   mI[16][160];

    const float* xb = x   + (long)b * sB;
    const float* sq = xsq + (long)b * N;

    // private sorted (descending) top-20 list
    float bv[20]; int bi[20];
#pragma unroll
    for (int t = 0; t < 20; ++t) { bv[t] = -3.4e38f; bi[t] = 0; }

    // squared norms of the accumulator rows owned by this lane
    float rsq[8];
#pragma unroll
    for (int e = 0; e < 8; ++e) rsq[e] = sq[r0 + hi * 8 + e];

#if __has_builtin(__builtin_amdgcn_wmma_f32_16x16x4_f32)
    // A fragment (16 rows x CP), register resident.
    // f32 A layout: lanes0-15: K=base+0/1, lanes16-31: K=base+2/3, M=lane&15.
    float a0[CP / 4], a1[CP / 4];
#pragma unroll
    for (int ks = 0; ks < CP / 4; ++ks) {
        int k0 = ks * 4 + hi * 2;
        int k1 = k0 + 1;
        if constexpr (CP > C) {
            int k0c = (k0 < C) ? k0 : 0;
            int k1c = (k1 < C) ? k1 : 0;
            float v0 = xb[(long)k0c * N + r0 + l15];
            float v1 = xb[(long)k1c * N + r0 + l15];
            a0[ks] = (k0 < C) ? v0 : 0.f;
            a1[ks] = (k1 < C) ? v1 : 0.f;
        } else {
            a0[ks] = xb[(long)k0 * N + r0 + l15];
            a1[ks] = xb[(long)k1 * N + r0 + l15];
        }
    }
#endif

    int myrow  = lane >> 1;        // row this lane scans (2 lanes per row)
    int mycol0 = (lane & 1) * 8;   // which half of the 16-col tile

    for (int t = 0; t < 64; ++t) {
        int col0 = (wv + 4 * t) * 16;
        v8f acc = {};
#if __has_builtin(__builtin_amdgcn_wmma_f32_16x16x4_f32)
#pragma unroll
        for (int ks = 0; ks < CP / 4; ++ks) {
            int k0 = ks * 4 + hi * 2;
            int k1 = k0 + 1;
            v2f av; av.x = a0[ks]; av.y = a1[ks];
            v2f bw;
            if constexpr (CP > C) {
                int k0c = (k0 < C) ? k0 : 0;
                int k1c = (k1 < C) ? k1 : 0;
                float v0 = xb[(long)k0c * N + col0 + l15];
                float v1 = xb[(long)k1c * N + col0 + l15];
                bw.x = (k0 < C) ? v0 : 0.f;
                bw.y = (k1 < C) ? v1 : 0.f;
            } else {
                bw.x = xb[(long)k0 * N + col0 + l15];
                bw.y = xb[(long)k1 * N + col0 + l15];
            }
            acc = __builtin_amdgcn_wmma_f32_16x16x4_f32(
                false, av, false, bw, (short)0, acc, false, false);
        }
#else
#pragma unroll
        for (int e = 0; e < 8; ++e) {
            int m = r0 + hi * 8 + e;
            float s = 0.f;
            for (int c = 0; c < C; ++c)
                s += xb[(long)c * N + m] * xb[(long)c * N + col0 + l15];
            acc[e] = s;
        }
#endif
        float cs = sq[col0 + l15];
#pragma unroll
        for (int e = 0; e < 8; ++e)
            tbuf[wv][hi * 8 + e][l15] = 2.f * acc[e] - rsq[e] - cs;
        __builtin_amdgcn_wave_barrier();   // DS ops are in-order within a wave

#pragma unroll
        for (int cc = 0; cc < 8; ++cc) {
            float v = tbuf[wv][myrow][mycol0 + cc];
            int  gi = col0 + mycol0 + cc;
            if (v > bv[19]) {
                bv[19] = v; bi[19] = gi;
#pragma unroll
                for (int s = 19; s > 0; --s) {
                    if (bv[s] > bv[s - 1]) {
                        float tv = bv[s]; bv[s] = bv[s - 1]; bv[s - 1] = tv;
                        int   ti = bi[s]; bi[s] = bi[s - 1]; bi[s - 1] = ti;
                    }
                }
            }
        }
        __builtin_amdgcn_wave_barrier();
    }

    int slot = wv * 2 + (lane & 1);   // 8 partial lists per row
#pragma unroll
    for (int s = 0; s < 20; ++s) {
        mV[myrow][slot * 20 + s] = bv[s];
        mI[myrow][slot * 20 + s] = bi[s];
    }
    __syncthreads();

    if (tid < 16) {   // 8-way merge of sorted lists; ties -> lower index
        int p[8] = {0, 0, 0, 0, 0, 0, 0, 0};
        for (int outk = 0; outk < KNN_K; ++outk) {
            float bestv = -3.4e38f; int besti = 0x7fffffff; int bests = 0;
            for (int s = 0; s < 8; ++s) {
                if (p[s] < 20) {
                    float v = mV[tid][s * 20 + p[s]];
                    int  ii = mI[tid][s * 20 + p[s]];
                    if (v > bestv || (v == bestv && ii < besti)) {
                        bestv = v; besti = ii; bests = s;
                    }
                }
            }
            p[bests]++;
            idxOut[((long)b * N + r0 + tid) * KNN_K + outk] = besti;
        }
    }
}

// ---------------------------------------------------------------------------
// Universal channel-mixing GEMM:  out[b][o][q] = sum_c W[o][c] * F[b][c][q]
// Fully specialized on <MODE, O, CIN, KP, CHALF, CSPLIT, OUTF32> so bounds
// guards fold away; padding handled with clamped loads + selects (no branches).
// 256 threads, 64x128 output tile, f16 WMMA 16x16x32, double-buffered LDS.
// ---------------------------------------------------------------------------
template<int MODE, int O, int CIN, int KP, int CHALF, int CSPLIT, bool OUTF32>
__global__ __launch_bounds__(256)
void gemm_conv(const float* __restrict__ W, int cols, int npts,
               const void* __restrict__ act, long actSB,
               const float* __restrict__ srcA,
               const float* __restrict__ srcB, long srcBSB,
               const float* __restrict__ xg, long xgSB,
               const int* __restrict__ idxp,
               void* __restrict__ out, long outSB)
{
    static_assert(MODE != MODE_F16 || KP == CIN, "dense f16 expects KP==CIN");
    __shared__ __align__(32) _Float16 Wl[2][64 * 32];   // [m][k], k contiguous
    __shared__ __align__(32) _Float16 Fl[2][128 * 32];  // [q][k], k contiguous

    const int b   = blockIdx.z;
    const int q0  = blockIdx.x * 128;
    const int o0  = blockIdx.y * 64;
    const int tid = threadIdx.x;

    // F-loader mapping: 2 threads per column, 16 channels each
    const int qf = tid >> 1;
    const int kf = (tid & 1) * 16;
    const int qg = q0 + qf;

    int nEdge = 0, jEdge = 0;
    if constexpr (MODE == MODE_EDGE) {
        nEdge = qg / KNN_K;
        jEdge = idxp[(long)b * cols + qg];
    }

    // wave -> subtile mapping
    const int wv = tid >> 5, lane = tid & 31;
    const int l15 = lane & 15, hi = lane >> 4;
    const int mt  = wv & 3;
    const int ntb = (wv >> 2) * 4;

    // ---- tile loaders (branch-free; guards are compile-time) ----
    auto loadW = [&](int kc, _Float16* dst) {
        const int m  = tid >> 2;
        const int kb = (tid & 3) * 8;
        const int o  = o0 + m;
        long orow;
        if constexpr (O % 64 == 0) orow = (long)o * CIN;
        else                       orow = (long)((o < O) ? o : 0) * CIN;
        _Float16* dp = dst + m * 32 + kb;
#pragma unroll
        for (int t = 0; t < 8; ++t) {
            int c = kc + kb + t;
            float v;
            if constexpr (KP == CIN) v = W[orow + c];
            else {
                int cc = (c < CIN) ? c : 0;
                v = W[orow + cc];
                v = (c < CIN) ? v : 0.f;
            }
            if constexpr (O % 64 != 0) v = (o < O) ? v : 0.f;
            dp[t] = (_Float16)v;
        }
    };

    auto loadF = [&](int kc, _Float16* dst) {
        _Float16* dp = dst + qf * 32 + kf;
        if constexpr (MODE == MODE_F16) {
            const _Float16* ap = (const _Float16*)act
                + (long)b * actSB + (long)(kc + kf) * cols + qg;
#pragma unroll
            for (int t = 0; t < 16; ++t) dp[t] = ap[(long)t * cols];
        } else if constexpr (MODE == MODE_F32) {
            const float* ap = (const float*)act
                + (long)b * actSB + (long)(kc + kf) * cols + qg;
#pragma unroll
            for (int t = 0; t < 16; ++t) dp[t] = (_Float16)ap[(long)t * cols];
        } else if constexpr (MODE == MODE_CAT) {
            if (kc + 32 <= CSPLIT) {   // block-uniform (CSPLIT % 32 == 0)
                const float* ap = srcA + (long)b * CSPLIT + kc + kf;
#pragma unroll
                for (int t = 0; t < 16; ++t) dp[t] = (_Float16)ap[t];
            } else {
                const float* ap = srcB + (long)b * srcBSB
                    + (long)(kc + kf - CSPLIT) * cols + qg;
#pragma unroll
                for (int t = 0; t < 16; ++t) dp[t] = (_Float16)ap[(long)t * cols];
            }
        } else { // MODE_EDGE: f = [x[:,j] - x[:,n], x[:,n]] (+ zero pad)
            const float* xb = xg + (long)b * xgSB;
#pragma unroll
            for (int t = 0; t < 16; ++t) {
                int  c      = kc + kf + t;
                bool isDiff = c < CHALF;
                int  cl     = isDiff ? c : (c - CHALF);
                if constexpr (KP > 2 * CHALF) cl = (cl < CHALF) ? cl : 0;
                long base = (long)cl * npts;
                float vn = xb[base + nEdge];
                float vj = xb[base + jEdge];
                float v  = isDiff ? (vj - vn) : vn;
                if constexpr (KP > 2 * CHALF) v = (c < 2 * CHALF) ? v : 0.f;
                dp[t] = (_Float16)v;
            }
        }
    };

    v8f acc0 = {}, acc1 = {}, acc2 = {}, acc3 = {};

    loadW(0, Wl[0]);
    loadF(0, Fl[0]);
    int cur = 0;

    for (int kc = 0; kc < KP; kc += 32) {
        __syncthreads();
        int nxt = cur ^ 1;
        if (kc + 32 < KP) {        // prefetch next tile while computing
            loadW(kc + 32, Wl[nxt]);
            loadF(kc + 32, Fl[nxt]);
        }
        // A fragment: K = hi*16..+15, contiguous (matches B)
        v16h afrag;
        {
            const _Float16* ap = &Wl[cur][(mt * 16 + l15) * 32 + hi * 16];
            v8h lo  = *(const v8h*)(ap);
            v8h hi8 = *(const v8h*)(ap + 8);
            afrag = __builtin_shufflevector(lo, hi8,
                     0,1,2,3,4,5,6,7,8,9,10,11,12,13,14,15);
        }
#pragma unroll
        for (int i = 0; i < 4; ++i) {
            const _Float16* bp = &Fl[cur][((ntb + i) * 16 + l15) * 32 + hi * 16];
            v16h bfrag = *(const v16h*)bp;   // K = hi*16..+15, contiguous 32B
            v8f& acc = (i == 0 ? acc0 : i == 1 ? acc1 : i == 2 ? acc2 : acc3);
            acc = __builtin_amdgcn_wmma_f32_16x16x32_f16(
                false, afrag, false, bfrag, (short)0, acc, false, false);
        }
        cur = nxt;
    }

    // ---- store (C/D layout: VGPR e -> M = hi*8+e, N = lane&15) ----
#pragma unroll
    for (int i = 0; i < 4; ++i) {
        v8f acc = (i == 0 ? acc0 : i == 1 ? acc1 : i == 2 ? acc2 : acc3);
        int qo = q0 + (ntb + i) * 16 + l15;
#pragma unroll
        for (int e = 0; e < 8; ++e) {
            int o = o0 + mt * 16 + hi * 8 + e;
            bool ok = true;
            if constexpr (O % 16 != 0) ok = (o < O);
            if (ok) {
                if constexpr (OUTF32)
                    ((float*)out)[(long)b * outSB + (long)o * cols + qo] = acc[e];
                else
                    ((_Float16*)out)[(long)b * outSB + (long)o * cols + qo] = (_Float16)acc[e];
            }
        }
    }
}

// ---------------------------------------------------------------------------
// BN stats, stage 1: deterministic partial sums per (channel, column chunk).
// part[o][ch] = {sum, sumsq} over b and q in chunk.
// ---------------------------------------------------------------------------
__global__ __launch_bounds__(256)
void bn_stats_partial(const _Float16* __restrict__ y, long sB, int cols,
                      int Bn, int chunk, int nchunks,
                      float* __restrict__ part)
{
    __shared__ float rs[256], rs2[256];
    int o = blockIdx.x, ch = blockIdx.y, tid = threadIdx.x;
    int q0 = ch * chunk;
    int q1 = q0 + chunk; if (q1 > cols) q1 = cols;
    float s = 0.f, s2 = 0.f;
    for (int b = 0; b < Bn; ++b) {
        const _Float16* p = y + (long)b * sB + (long)o * cols;
        for (int q = q0 + tid; q < q1; q += 256) {
            float v = (float)p[q];
            s += v; s2 += v * v;
        }
    }
    rs[tid] = s; rs2[tid] = s2;
    __syncthreads();
    for (int st = 128; st > 0; st >>= 1) {
        if (tid < st) { rs[tid] += rs[tid + st]; rs2[tid] += rs2[tid + st]; }
        __syncthreads();
    }
    if (tid == 0) {
        part[((long)o * nchunks + ch) * 2]     = rs[0];
        part[((long)o * nchunks + ch) * 2 + 1] = rs2[0];
    }
}

// BN stats, stage 2: reduce chunks, fold gamma/beta into per-channel affine.
__global__ __launch_bounds__(64)
void bn_stats_final(const float* __restrict__ part, int nchunks, float cnt,
                    const float* __restrict__ g, const float* __restrict__ beta,
                    float* __restrict__ ab)
{
    __shared__ float rs[64], rs2[64];
    int o = blockIdx.x, tid = threadIdx.x;
    float s = 0.f, s2 = 0.f;
    for (int c = tid; c < nchunks; c += 64) {
        s  += part[((long)o * nchunks + c) * 2];
        s2 += part[((long)o * nchunks + c) * 2 + 1];
    }
    rs[tid] = s; rs2[tid] = s2;
    __syncthreads();
    for (int st = 32; st > 0; st >>= 1) {
        if (tid < st) { rs[tid] += rs[tid + st]; rs2[tid] += rs2[tid + st]; }
        __syncthreads();
    }
    if (tid == 0) {
        float mean = rs[0] / cnt;
        float var  = rs2[0] / cnt - mean * mean;
        float istd = rsqrtf(var + 1e-5f);
        float a = g[o] * istd;
        ab[2 * o]     = a;
        ab[2 * o + 1] = beta[o] - mean * a;
    }
}

// normalize + LReLU in place on f16 buffer, 8 elements per thread
__global__ __launch_bounds__(256)
void bn_lrelu_kernel(_Float16* __restrict__ y, const float* __restrict__ ab,
                     int O, int cols, long total8)
{
    long i = (long)blockIdx.x * 256 + threadIdx.x;
    if (i >= total8) return;
    long e0 = i * 8;
    int o = (int)((e0 / cols) % O);
    float a = ab[2 * o], c = ab[2 * o + 1];
    v8h v = *(v8h*)(y + e0);
    v8h r;
#pragma unroll
    for (int t = 0; t < 8; ++t) r[t] = (_Float16)lrelu((float)v[t] * a + c);
    *(v8h*)(y + e0) = r;
}

// normalize + LReLU + max over K neighbors -> f32 slot of x123
__global__ __launch_bounds__(256)
void bn_lrelu_maxk_kernel(const _Float16* __restrict__ y,
                          const float* __restrict__ ab, int O, int npts,
                          float* __restrict__ dst, long dstSB, long total)
{
    long i = (long)blockIdx.x * 256 + threadIdx.x;
    if (i >= total) return;
    int n = (int)(i % npts);
    int o = (int)((i / npts) % O);
    int b = (int)(i / ((long)npts * O));
    const _Float16* p = y + ((long)b * O + o) * ((long)npts * KNN_K) + (long)n * KNN_K;
    float a = ab[2 * o], c = ab[2 * o + 1];
    float m = -3.4e38f;
#pragma unroll
    for (int k = 0; k < KNN_K; ++k) {
        float v = lrelu((float)p[k] * a + c);
        m = fmaxf(m, v);
    }
    dst[(long)b * dstSB + (long)o * npts + n] = m;
}

// normalize + LReLU + max over n -> gmax[b][o]
__global__ __launch_bounds__(256)
void bn_lrelu_maxn_kernel(const _Float16* __restrict__ y,
                          const float* __restrict__ ab, int O, int cols,
                          float* __restrict__ gmax)
{
    __shared__ float red[256];
    int o = blockIdx.x, b = blockIdx.y, tid = threadIdx.x;
    const _Float16* p = y + ((long)b * O + o) * cols;
    float a = ab[2 * o], c = ab[2 * o + 1];
    float m = -3.4e38f;
    for (int n = tid; n < cols; n += 256)
        m = fmaxf(m, lrelu((float)p[n] * a + c));
    red[tid] = m;
    __syncthreads();
    for (int st = 128; st > 0; st >>= 1) {
        if (tid < st) red[tid] = fmaxf(red[tid], red[tid + st]);
        __syncthreads();
    }
    if (tid == 0) gmax[(long)b * O + o] = red[0];
}

// ---------------------------------------------------------------------------
extern "C" void kernel_launch(void* const* d_in, const int* in_sizes, int n_in,
                              void* d_out, int out_size, void* d_ws, size_t ws_size,
                              hipStream_t stream)
{
    const float* x   = (const float*)d_in[0];
    const float* w1a = (const float*)d_in[1];
    const float* g1a = (const float*)d_in[2];  const float* b1a = (const float*)d_in[3];
    const float* w1b = (const float*)d_in[4];
    const float* g1b = (const float*)d_in[5];  const float* b1b = (const float*)d_in[6];
    const float* w2a = (const float*)d_in[7];
    const float* g2a = (const float*)d_in[8];  const float* b2a = (const float*)d_in[9];
    const float* w2b = (const float*)d_in[10];
    const float* g2b = (const float*)d_in[11]; const float* b2b = (const float*)d_in[12];
    const float* w3  = (const float*)d_in[13];
    const float* g3  = (const float*)d_in[14]; const float* b3  = (const float*)d_in[15];
    const float* w4  = (const float*)d_in[16];
    const float* g4  = (const float*)d_in[17]; const float* b4  = (const float*)d_in[18];
    const float* w5  = (const float*)d_in[19];
    const float* g5  = (const float*)d_in[20]; const float* b5  = (const float*)d_in[21];
    const float* w6  = (const float*)d_in[22];
    const float* g6  = (const float*)d_in[23]; const float* b6  = (const float*)d_in[24];
    const float* w7  = (const float*)d_in[25];
    const float* g7  = (const float*)d_in[26]; const float* b7  = (const float*)d_in[27];
    const float* w8  = (const float*)d_in[28];

    const int  N  = NPTS;
    const int  B  = NB;
    const long sE = (long)64 * ECOLS;        // edge tensor batch stride (elems)

    // ---- workspace layout (~252 MB) ----
    char* ws = (char*)d_ws;
    int*      idx  = (int*)(ws + 0);                          //  2.62 MB
    float*    xsq  = (float*)(ws + 2621440);                  //  0.13 MB
    float*    ab   = (float*)(ws + 2752512);                  //  8 KB
    float*    part = (float*)(ws + 2760704);                  //  <= 20 KB
    _Float16* e0   = (_Float16*)(ws + 3145728);               // 84 MB
    _Float16* e1   = (_Float16*)(ws + 3145728 + 83886080);
    float*    x123 = (float*)(ws + 87031808 + 83886080);      // 24 MB f32
    _Float16* h4   = (_Float16*)(ws + 170917888 + 25165824);  // 64 MB
    float*    gmax = (float*)(ws + 196083712 + 67108864);
    _Float16* h5   = e0;                                      // reuse
    _Float16* h6   = e1;                                      // reuse
    _Float16* h7   = (_Float16*)((char*)e0 + 33554432);       // reuse tail

    dim3 blk256(256), blk128(128), blk64(64);
    dim3 gEdge(ECOLS / 128, 1, B);
    long totE  = (long)B * 64 * ECOLS;
    int  gbE8  = (int)((totE / 8 + 255) / 256);
    long totMK = (long)B * 64 * N;
    int  gbMK  = (int)((totMK + 255) / 256);

    const int nchE = ECOLS / STCH;   // 40 chunks for edge tensors
    const int nchN = N / STCH;       // 2 chunks for point tensors
    const float cntE = (float)B * (float)ECOLS;
    const float cntN = (float)B * (float)N;

    auto statsE = [&](const _Float16* y, const float* g, const float* be) {
        bn_stats_partial<<<dim3(64, nchE), blk256, 0, stream>>>(y, sE, ECOLS, B, STCH, nchE, part);
        bn_stats_final<<<64, blk64, 0, stream>>>(part, nchE, cntE, g, be, ab);
    };

    // ===== EdgeConv block 1 (C=6) =====
    sqnorm_kernel<6><<<dim3(N / 256, B), blk256, 0, stream>>>(x, 6L * N, N, xsq);
    knn_kernel<8, 6><<<dim3(N / 16, B), blk128, 0, stream>>>(x, 6L * N, N, xsq, idx);

    gemm_conv<MODE_EDGE, 64, 12, 32, 6, 0, false><<<gEdge, blk256, 0, stream>>>(
        w1a, ECOLS, N, nullptr, 0, nullptr, nullptr, 0, x, 6L * N, idx, e0, sE);
    statsE(e0, g1a, b1a);
    bn_lrelu_kernel<<<gbE8, blk256, 0, stream>>>(e0, ab, 64, ECOLS, totE / 8);

    gemm_conv<MODE_F16, 64, 64, 64, 0, 0, false><<<gEdge, blk256, 0, stream>>>(
        w1b, ECOLS, N, e0, sE, nullptr, nullptr, 0, nullptr, 0, nullptr, e1, sE);
    statsE(e1, g1b, b1b);
    bn_lrelu_maxk_kernel<<<gbMK, blk256, 0, stream>>>(e1, ab, 64, N,
        x123, 192L * N, totMK);                               // -> x1 (chans 0..63)

    // ===== EdgeConv block 2 (C=64) =====
    sqnorm_kernel<64><<<dim3(N / 256, B), blk256, 0, stream>>>(x123, 192L * N, N, xsq);
    knn_kernel<64, 64><<<dim3(N / 16, B), blk128, 0, stream>>>(x123, 192L * N, N, xsq, idx);

    gemm_conv<MODE_EDGE, 64, 128, 128, 64, 0, false><<<gEdge, blk256, 0, stream>>>(
        w2a, ECOLS, N, nullptr, 0, nullptr, nullptr, 0, x123, 192L * N, idx, e0, sE);
    statsE(e0, g2a, b2a);
    bn_lrelu_kernel<<<gbE8, blk256, 0, stream>>>(e0, ab, 64, ECOLS, totE / 8);

    gemm_conv<MODE_F16, 64, 64, 64, 0, 0, false><<<gEdge, blk256, 0, stream>>>(
        w2b, ECOLS, N, e0, sE, nullptr, nullptr, 0, nullptr, 0, nullptr, e1, sE);
    statsE(e1, g2b, b2b);
    bn_lrelu_maxk_kernel<<<gbMK, blk256, 0, stream>>>(e1, ab, 64, N,
        x123 + 64L * N, 192L * N, totMK);                     // -> x2 (chans 64..127)

    // ===== EdgeConv block 3 (C=64, single conv) =====
    sqnorm_kernel<64><<<dim3(N / 256, B), blk256, 0, stream>>>(x123 + 64L * N, 192L * N, N, xsq);
    knn_kernel<64, 64><<<dim3(N / 16, B), blk128, 0, stream>>>(x123 + 64L * N, 192L * N, N, xsq, idx);

    gemm_conv<MODE_EDGE, 64, 128, 128, 64, 0, false><<<gEdge, blk256, 0, stream>>>(
        w3, ECOLS, N, nullptr, 0, nullptr, nullptr, 0, x123 + 64L * N, 192L * N, idx, e0, sE);
    statsE(e0, g3, b3);
    bn_lrelu_maxk_kernel<<<gbMK, blk256, 0, stream>>>(e0, ab, 64, N,
        x123 + 128L * N, 192L * N, totMK);                    // -> x3 (chans 128..191)

    // ===== conv4: 192 -> 1024 over x123, then global max over n =====
    gemm_conv<MODE_F32, 1024, 192, 192, 0, 0, false><<<dim3(N / 128, 16, B), blk256, 0, stream>>>(
        w4, N, N, x123, 192L * N, nullptr, nullptr, 0, nullptr, 0, nullptr, h4, 1024L * N);
    bn_stats_partial<<<dim3(1024, nchN), blk256, 0, stream>>>(h4, 1024L * N, N, B, STCH, nchN, part);
    bn_stats_final<<<1024, blk64, 0, stream>>>(part, nchN, cntN, g4, b4, ab);
    bn_lrelu_maxn_kernel<<<dim3(1024, B), blk256, 0, stream>>>(h4, ab, 1024, N, gmax);

    // ===== conv5: concat(gmax, x123) 1216 -> 256 =====
    gemm_conv<MODE_CAT, 256, 1216, 1216, 0, 1024, false><<<dim3(N / 128, 4, B), blk256, 0, stream>>>(
        w5, N, N, nullptr, 0, gmax, x123, 192L * N, nullptr, 0, nullptr, h5, 256L * N);
    long tot5 = (long)B * 256 * N; int gb58 = (int)((tot5 / 8 + 255) / 256);
    bn_stats_partial<<<dim3(256, nchN), blk256, 0, stream>>>(h5, 256L * N, N, B, STCH, nchN, part);
    bn_stats_final<<<256, blk64, 0, stream>>>(part, nchN, cntN, g5, b5, ab);
    bn_lrelu_kernel<<<gb58, blk256, 0, stream>>>(h5, ab, 256, N, tot5 / 8);

    // ===== conv6: 256 -> 256 =====
    gemm_conv<MODE_F16, 256, 256, 256, 0, 0, false><<<dim3(N / 128, 4, B), blk256, 0, stream>>>(
        w6, N, N, h5, 256L * N, nullptr, nullptr, 0, nullptr, 0, nullptr, h6, 256L * N);
    bn_stats_partial<<<dim3(256, nchN), blk256, 0, stream>>>(h6, 256L * N, N, B, STCH, nchN, part);
    bn_stats_final<<<256, blk64, 0, stream>>>(part, nchN, cntN, g6, b6, ab);
    bn_lrelu_kernel<<<gb58, blk256, 0, stream>>>(h6, ab, 256, N, tot5 / 8);

    // ===== conv7: 256 -> 128 =====
    gemm_conv<MODE_F16, 128, 256, 256, 0, 0, false><<<dim3(N / 128, 2, B), blk256, 0, stream>>>(
        w7, N, N, h6, 256L * N, nullptr, nullptr, 0, nullptr, 0, nullptr, h7, 128L * N);
    long tot7 = (long)B * 128 * N; int gb78 = (int)((tot7 / 8 + 255) / 256);
    bn_stats_partial<<<dim3(128, nchN), blk256, 0, stream>>>(h7, 128L * N, N, B, STCH, nchN, part);
    bn_stats_final<<<128, blk64, 0, stream>>>(part, nchN, cntN, g7, b7, ab);
    bn_lrelu_kernel<<<gb78, blk256, 0, stream>>>(h7, ab, 128, N, tot7 / 8);

    // ===== conv8: 128 -> 6 (no BN), f32 output =====
    gemm_conv<MODE_F16, 6, 128, 128, 0, 0, true><<<dim3(N / 128, 1, B), blk256, 0, stream>>>(
        w8, N, N, h7, 128L * N, nullptr, nullptr, 0, nullptr, 0, nullptr, (float*)d_out, 6L * N);

    (void)in_sizes; (void)n_in; (void)out_size; (void)ws_size;
}
